// Model_28879360098358
// MI455X (gfx1250) — matrix operand order repaired and checked by reading, not compiled, not run on hardware
//
#include <hip/hip_runtime.h>
#include <hip/hip_bf16.h>

// ---------------------------------------------------------------------------
// MI455X (gfx1250) implementation.
//  - All big contractions run through one generic batched bf16 WMMA GEMM
//    (v_wmma_f32_16x16x32_bf16, f32 accumulate), tiles staged in LDS.
//  - 64x128x64 block tile, 8 waves (2x4), 32x32 per wave -> 8 WMMAs per
//    staging barrier with 2x fragment reuse (fa x fb outer product).
//  - Fast staging path: async global->LDS (ASYNCcnt) when the toolchain
//    exposes it, else branch-free b64 global->LDS copies; plus
//    global_prefetch of the next K tile. Edge tiles use clamped-address +
//    select (no exec divergence).
//  - bf16 chosen over f16: inputs are f32, bf16 preserves exponent range;
//    accumulation stays f32 so softmax/norm statistics remain sane.
// ---------------------------------------------------------------------------

typedef __attribute__((ext_vector_type(16))) __bf16 v16bf;
typedef __attribute__((ext_vector_type(8)))  float  v8f;

#define TB 256
#define BM 64
#define BN 128
#define BK 64
#define LDK (BK + 4)   // row stride in ushorts; 136B, 8B-aligned

#if __has_builtin(__builtin_amdgcn_global_load_async_to_lds_b64) && \
    __has_builtin(__builtin_amdgcn_s_wait_asynccnt)
#define HAS_ASYNC_LDS 1
typedef int v2i __attribute__((ext_vector_type(2)));
typedef __attribute__((address_space(1))) v2i* as1_v2i_p;   // global b64 payload ptr
typedef __attribute__((address_space(3))) v2i* as3_v2i_p;   // LDS b64 payload ptr
#else
#define HAS_ASYNC_LDS 0
#endif

__device__ inline float bf2f(unsigned short h) {
  union { unsigned u; float f; } x; x.u = ((unsigned)h) << 16; return x.f;
}
__device__ inline unsigned short f2bf(float f) {
  union { float f; unsigned u; } x; x.f = f;
  unsigned r = x.u + 0x7FFFu + ((x.u >> 16) & 1u);
  return (unsigned short)(r >> 16);
}
__device__ inline float gelu_exact(float v) {
  return 0.5f * v * (1.0f + erff(v * 0.70710678118654752f));
}
__device__ inline float block_reduce_sum(float v) {
  __shared__ float red[TB];
  int t = threadIdx.x; red[t] = v; __syncthreads();
  for (int s = TB / 2; s > 0; s >>= 1) { if (t < s) red[t] += red[t + s]; __syncthreads(); }
  float r = red[0]; __syncthreads(); return r;
}
__device__ inline float block_reduce_max(float v) {
  __shared__ float red[TB];
  int t = threadIdx.x; red[t] = v; __syncthreads();
  for (int s = TB / 2; s > 0; s >>= 1) { if (t < s) red[t] = fmaxf(red[t], red[t + s]); __syncthreads(); }
  float r = red[0]; __syncthreads(); return r;
}
__device__ inline float hash_u01(unsigned x) {
  x ^= x >> 17; x *= 0xED5AD4BBu; x ^= x >> 11; x *= 0xAC4C1B51u;
  x ^= x >> 15; x *= 0x31848BABu; x ^= x >> 14;
  return (float)(x >> 8) * (1.0f / 16777216.0f);
}

// ---------------------------------------------------------------------------
// Generic batched bf16 WMMA GEMM:  D = resid + act(alpha*colscale*(A@B) + bias)
// Fragment layouts per CDNA5 ISA 7.12.2 (16-bit A 16x32, 16-bit B 32x16).
// ---------------------------------------------------------------------------
__global__ __launch_bounds__(TB) void k_gemm_bf16_wmma(
    const unsigned short* __restrict__ A, int lda, long long sA, int transA,
    const unsigned short* __restrict__ Bw, int ldb, long long sB, int transB,
    float* __restrict__ D, unsigned short* __restrict__ Dbf,
    int ldd, long long sD, int storeT,
    const float* __restrict__ bias, long long sBias,
    const float* __restrict__ colscale,
    const float* __restrict__ resid, long long sRes,
    float alpha, int act, int M, int N, int Kd)
{
  __shared__ unsigned short As[BM][LDK];
  __shared__ unsigned short Bs[BN][LDK];   // stored [n][k]

  const int bz = blockIdx.z;
  A  += (long long)bz * sA;
  Bw += (long long)bz * sB;
  float*          Dp  = D   ? D   + (long long)bz * sD : nullptr;
  unsigned short* Dbp = Dbf ? Dbf + (long long)bz * sD : nullptr;
  const float*    bp  = bias  ? bias  + (long long)bz * sBias : nullptr;
  const float*    rp  = resid ? resid + (long long)bz * sRes  : nullptr;

  const int m0 = blockIdx.y * BM, n0 = blockIdx.x * BN;
  const int tid = threadIdx.x, lane = tid & 31, wave = tid >> 5;
  const int wrow = (wave >> 2) * 32, wcol = (wave & 3) * 32;
  const int h = lane >> 4, ml = lane & 15;

  v8f acc[2][2] = {};

  const bool aContig = (transA == 0) && ((lda & 3) == 0);
  const bool bContig = (transB != 0) && ((ldb & 3) == 0);
  const bool fullM = (m0 + BM) <= M;
  const bool fullN = (n0 + BN) <= N;

  for (int k0 = 0; k0 < Kd; k0 += BK) {
    const bool fullK = (k0 + BK) <= Kd;

    // ---- stage A tile [BM][BK] ----
    if (aContig && fullM && fullK) {
      // branch-free path: 8B per thread-iter, contiguous in K
      for (int i = tid; i < BM * BK / 4; i += TB) {
        int r = i >> 4, c4 = (i & 15) * 4;
        const unsigned long long* src =
            (const unsigned long long*)(A + (long long)(m0 + r) * lda + k0 + c4);
#if HAS_ASYNC_LDS
        __builtin_amdgcn_global_load_async_to_lds_b64(
            (as1_v2i_p)(unsigned long long)src,
            (as3_v2i_p)(unsigned)(unsigned long long)&As[r][c4], 0, 0);
#else
        *(unsigned long long*)&As[r][c4] = *src;
#endif
        __builtin_prefetch((const void*)((const char*)src + BK * 2), 0, 1);
      }
    } else {
      for (int i = tid; i < BM * BK; i += TB) {
        int r = i >> 6, c = i & (BK - 1);
        int gm = m0 + r, gk = k0 + c;
        int cm = gm < M ? gm : M - 1;
        int ck = gk < Kd ? gk : Kd - 1;
        unsigned short v = transA ? A[(long long)ck * lda + cm]
                                  : A[(long long)cm * lda + ck];
        As[r][c] = (gm < M && gk < Kd) ? v : (unsigned short)0;
      }
    }

    // ---- stage B tile [BN][BK] (k-major rows) ----
    if (bContig && fullN && fullK) {
      for (int i = tid; i < BN * BK / 4; i += TB) {
        int r = i >> 4, c4 = (i & 15) * 4;
        const unsigned long long* src =
            (const unsigned long long*)(Bw + (long long)(n0 + r) * ldb + k0 + c4);
#if HAS_ASYNC_LDS
        __builtin_amdgcn_global_load_async_to_lds_b64(
            (as1_v2i_p)(unsigned long long)src,
            (as3_v2i_p)(unsigned)(unsigned long long)&Bs[r][c4], 0, 0);
#else
        *(unsigned long long*)&Bs[r][c4] = *src;
#endif
        __builtin_prefetch((const void*)((const char*)src + BK * 2), 0, 1);
      }
    } else {
      for (int i = tid; i < BN * BK; i += TB) {
        int r = i >> 6, c = i & (BK - 1);
        int gn = n0 + r, gk = k0 + c;
        int cn = gn < N ? gn : N - 1;
        int ck = gk < Kd ? gk : Kd - 1;
        unsigned short v = transB ? Bw[(long long)cn * ldb + ck]
                                  : Bw[(long long)ck * ldb + cn];
        Bs[r][c] = (gn < N && gk < Kd) ? v : (unsigned short)0;
      }
    }
#if HAS_ASYNC_LDS
    __builtin_amdgcn_s_wait_asynccnt(0);
#endif
    __syncthreads();

#pragma unroll
    for (int ks = 0; ks < BK; ks += 32) {
      union Frag { v16bf v; unsigned short u[16]; } fa0, fa1, fb0, fb1;
#pragma unroll
      for (int e = 0; e < 16; ++e) {            // A 16x32: k = h*8 + (e&7) + ((e&8)<<1)
        int kk = ks + h * 8 + (e & 7) + ((e & 8) << 1);
        fa0.u[e] = As[wrow + ml][kk];
        fa1.u[e] = As[wrow + 16 + ml][kk];
      }
#pragma unroll
      for (int e = 0; e < 16; ++e) {            // B 32x16: k = h*8 + (e&7) + ((e&8)<<1), n = ml
        int kk = ks + h * 8 + (e & 7) + ((e & 8) << 1);
        fb0.u[e] = Bs[wcol + ml][kk];
        fb1.u[e] = Bs[wcol + 16 + ml][kk];
      }
      acc[0][0] = __builtin_amdgcn_wmma_f32_16x16x32_bf16(false, fa0.v, false, fb0.v,
                                                          (short)0, acc[0][0], false, false);
      acc[0][1] = __builtin_amdgcn_wmma_f32_16x16x32_bf16(false, fa0.v, false, fb1.v,
                                                          (short)0, acc[0][1], false, false);
      acc[1][0] = __builtin_amdgcn_wmma_f32_16x16x32_bf16(false, fa1.v, false, fb0.v,
                                                          (short)0, acc[1][0], false, false);
      acc[1][1] = __builtin_amdgcn_wmma_f32_16x16x32_bf16(false, fa1.v, false, fb1.v,
                                                          (short)0, acc[1][1], false, false);
    }
    __syncthreads();
  }

#pragma unroll
  for (int mt = 0; mt < 2; ++mt) {
#pragma unroll
    for (int r = 0; r < 8; ++r) {               // C/D: m = r + 8*h, n = ml
      int mm = m0 + wrow + mt * 16 + r + h * 8;
      if (mm >= M) continue;
#pragma unroll
      for (int nt = 0; nt < 2; ++nt) {
        int nn = n0 + wcol + nt * 16 + ml;
        if (nn >= N) continue;
        float v = acc[mt][nt][r] * alpha;
        if (colscale) v *= colscale[nn];
        if (bp)       v += bp[nn];
        if (act == 1)      v = fmaxf(v, 0.0f);
        else if (act == 2) v = gelu_exact(v);
        long long idx = storeT ? (long long)nn * ldd + mm : (long long)mm * ldd + nn;
        if (rp)  v += rp[idx];
        if (Dp)  Dp[idx]  = v;
        if (Dbp) Dbp[idx] = f2bf(v);
      }
    }
  }
}

// ------------------------------ vector kernels ------------------------------
__global__ void k_f32_to_bf16(const float* __restrict__ in,
                              unsigned short* __restrict__ out, long long n) {
  long long i = (long long)blockIdx.x * blockDim.x + threadIdx.x;
  if (i < n) out[i] = f2bf(in[i]);
}

__global__ __launch_bounds__(TB) void k_l2norm_rows_bf16(
    const float* __restrict__ in, unsigned short* __restrict__ out, int D) {
  long long row = blockIdx.x;
  const float* p = in + row * (long long)D;
  float ss = 0.0f;
  for (int i = threadIdx.x; i < D; i += TB) { float v = p[i]; ss += v * v; }
  ss = block_reduce_sum(ss);
  float sc = rsqrtf(fmaxf(ss, 1e-24f));        // 1/clip(norm,1e-12)
  for (int i = threadIdx.x; i < D; i += TB) out[row * (long long)D + i] = f2bf(p[i] * sc);
}

__global__ __launch_bounds__(TB) void k_row_invnorm(
    const float* __restrict__ E, float* __restrict__ inv, int D) {
  int row = blockIdx.x;
  const float* p = E + (long long)row * D;
  float ss = 0.0f;
  for (int i = threadIdx.x; i < D; i += TB) { float v = p[i]; ss += v * v; }
  ss = block_reduce_sum(ss);
  if (threadIdx.x == 0) inv[row] = rsqrtf(fmaxf(ss, 1e-30f));
}

__global__ __launch_bounds__(TB) void k_softmax_rows_bf16(
    const float* __restrict__ X, unsigned short* __restrict__ out, int N) {
  long long row = blockIdx.x;
  const float* p = X + row * (long long)N;
  float mx = -3.4e38f;
  for (int i = threadIdx.x; i < N; i += TB) mx = fmaxf(mx, p[i]);
  mx = block_reduce_max(mx);
  float sm = 0.0f;
  for (int i = threadIdx.x; i < N; i += TB) sm += __expf(p[i] - mx);
  sm = block_reduce_sum(sm);
  float inv = 1.0f / sm;
  for (int i = threadIdx.x; i < N; i += TB)
    out[row * (long long)N + i] = f2bf(__expf(p[i] - mx) * inv);
}

// Cm = l2norm( (A@H) * Bernoulli(P)^T ), hash-based deterministic Bernoulli.
__global__ __launch_bounds__(TB) void k_cm_mask_l2norm(
    const float* __restrict__ Cmr, const unsigned short* __restrict__ Pbf,
    float* __restrict__ out, int Kc, int Sdim, int HD) {
  int b = blockIdx.x / Kc, k = blockIdx.x % Kc;
  const float* src = Cmr + (long long)blockIdx.x * HD;
  float ss = 0.0f;
  for (int hh = threadIdx.x; hh < HD; hh += TB) {
    float pch = bf2f(Pbf[((long long)b * Sdim + hh) * Kc + k]);
    unsigned seed = (unsigned)(b * 73856093) ^ (unsigned)(hh * 19349663) ^
                    (unsigned)(k * 83492791) ^ 0x9E3779B9u;
    float m = (hash_u01(seed) < pch) ? 1.0f : 0.0f;
    float v = src[hh] * m; ss += v * v;
  }
  ss = block_reduce_sum(ss);
  float sc = rsqrtf(fmaxf(ss, 1e-24f));
  for (int hh = threadIdx.x; hh < HD; hh += TB) {
    float pch = bf2f(Pbf[((long long)b * Sdim + hh) * Kc + k]);
    unsigned seed = (unsigned)(b * 73856093) ^ (unsigned)(hh * 19349663) ^
                    (unsigned)(k * 83492791) ^ 0x9E3779B9u;
    float m = (hash_u01(seed) < pch) ? 1.0f : 0.0f;
    out[(long long)blockIdx.x * HD + hh] = src[hh] * m * sc;
  }
}

__global__ void k_mul_wout_bf16(const float* __restrict__ theta,
                                const float* __restrict__ Wout,
                                unsigned short* __restrict__ out,
                                int Cn, int HDn, long long n) {
  long long i = (long long)blockIdx.x * blockDim.x + threadIdx.x;
  if (i >= n) return;
  int hh = (int)(i % HDn);
  int c  = (int)((i / HDn) % Cn);
  out[i] = f2bf(theta[i] * Wout[(long long)c * HDn + hh]);
}

__global__ void k_revin_stats(const float* __restrict__ Y, float* __restrict__ mu,
                              float* __restrict__ sd, int Bn, int Sn, int Cn) {
  int idx = blockIdx.x * blockDim.x + threadIdx.x;
  if (idx >= Bn * Cn) return;
  int b = idx / Cn, c = idx % Cn;
  const float* p = Y + (long long)b * Sn * Cn + c;
  float s = 0.0f, ss = 0.0f;
  for (int t = 0; t < Sn; ++t) { float v = p[(long long)t * Cn]; s += v; ss += v * v; }
  float m = s / Sn, var = ss / Sn - m * m;
  mu[idx] = m; sd[idx] = sqrtf(fmaxf(var, 0.0f) + 1e-5f);
}

__global__ void k_revin_apply(const float* __restrict__ Y, const float* __restrict__ mu,
                              const float* __restrict__ sd, const float* __restrict__ rw,
                              const float* __restrict__ rb, float* __restrict__ Z,
                              int Sn, int Cn, long long n) {
  long long i = (long long)blockIdx.x * blockDim.x + threadIdx.x;
  if (i >= n) return;
  int c = (int)(i % Cn);
  int b = (int)(i / ((long long)Cn * Sn));
  int bc = b * Cn + c;
  Z[i] = (Y[i] - mu[bc]) / sd[bc] * rw[c] + rb[c];
}

__global__ __launch_bounds__(TB) void k_bn_stats_s(const float* __restrict__ Z,
                                                   float* __restrict__ m, float* __restrict__ istd,
                                                   int Bn, int Sn, int Cn) {
  int s = blockIdx.x;
  float sm = 0.0f, ss = 0.0f;
  for (int i = threadIdx.x; i < Bn * Cn; i += TB) {
    int b = i / Cn, c = i % Cn;
    float v = Z[((long long)b * Sn + s) * Cn + c];
    sm += v; ss += v * v;
  }
  sm = block_reduce_sum(sm);
  ss = block_reduce_sum(ss);
  if (threadIdx.x == 0) {
    float cnt = (float)(Bn * Cn);
    float mean = sm / cnt, var = ss / cnt - mean * mean;
    m[s] = mean; istd[s] = rsqrtf(fmaxf(var, 0.0f) + 1e-5f);
  }
}

__global__ __launch_bounds__(TB) void k_bn_stats_c(const float* __restrict__ Z,
                                                   float* __restrict__ m, float* __restrict__ istd,
                                                   int Bn, int Sn, int Cn) {
  int c = blockIdx.x;
  float sm = 0.0f, ss = 0.0f;
  for (int i = threadIdx.x; i < Bn * Sn; i += TB) {
    int b = i / Sn, s = i % Sn;
    float v = Z[((long long)b * Sn + s) * Cn + c];
    sm += v; ss += v * v;
  }
  sm = block_reduce_sum(sm);
  ss = block_reduce_sum(ss);
  if (threadIdx.x == 0) {
    float cnt = (float)(Bn * Sn);
    float mean = sm / cnt, var = ss / cnt - mean * mean;
    m[c] = mean; istd[c] = rsqrtf(fmaxf(var, 0.0f) + 1e-5f);
  }
}

__global__ void k_bn_apply_s(const float* __restrict__ Z, const float* __restrict__ m,
                             const float* __restrict__ istd, const float* __restrict__ g,
                             const float* __restrict__ bb, unsigned short* __restrict__ out,
                             int Sn, int Cn, long long n) {
  long long i = (long long)blockIdx.x * blockDim.x + threadIdx.x;
  if (i >= n) return;
  int s = (int)((i / Cn) % Sn);
  out[i] = f2bf((Z[i] - m[s]) * istd[s] * g[s] + bb[s]);
}

__global__ void k_bn_apply_c(const float* __restrict__ Z, const float* __restrict__ m,
                             const float* __restrict__ istd, const float* __restrict__ g,
                             const float* __restrict__ bb, unsigned short* __restrict__ out,
                             int Sn, int Cn, long long n) {
  long long i = (long long)blockIdx.x * blockDim.x + threadIdx.x;
  if (i >= n) return;
  int c = (int)(i % Cn);
  out[i] = f2bf((Z[i] - m[c]) * istd[c] * g[c] + bb[c]);
}

__global__ void k_transpose_bsc_bcs(const float* __restrict__ Z,
                                    unsigned short* __restrict__ out,
                                    int Sn, int Cn, long long n) {
  long long i = (long long)blockIdx.x * blockDim.x + threadIdx.x;  // i -> [b][c][s]
  if (i >= n) return;
  int s = (int)(i % Sn);
  long long t = i / Sn;
  int c = (int)(t % Cn);
  int b = (int)(t / Cn);
  out[i] = f2bf(Z[((long long)b * Sn + s) * Cn + c]);
}

__global__ void k_denorm_out(const float* __restrict__ Otmp, const float* __restrict__ rw,
                             const float* __restrict__ rb, const float* __restrict__ mu,
                             const float* __restrict__ sd, float* __restrict__ out,
                             int Bn, int PLn, int Cn) {
  long long n = (long long)Bn * PLn * Cn;
  long long i = (long long)blockIdx.x * blockDim.x + threadIdx.x;  // i -> [b][p][c]
  if (i >= n) return;
  int c = (int)(i % Cn);
  long long t = i / Cn;
  int p = (int)(t % PLn);
  int b = (int)(t / PLn);
  float v = Otmp[((long long)c * Bn + b) * PLn + p];
  v = (v - rb[c]) / (rw[c] + 1e-10f) * sd[b * Cn + c] + mu[b * Cn + c];
  out[i] = v;
}

// ------------------------------ host orchestration ------------------------------
static inline unsigned cdivu(long long a, long long b) { return (unsigned)((a + b - 1) / b); }

static void gemm(hipStream_t st,
                 const unsigned short* A, int lda, long long sA, int tA,
                 const unsigned short* Bw, int ldb, long long sB, int tB,
                 float* D, unsigned short* Dbf, int ldd, long long sD, int storeT,
                 const float* bias, long long sBias, const float* colscale,
                 const float* resid, long long sRes, float alpha, int act,
                 int M, int N, int Kd, int batch) {
  dim3 g(cdivu(N, BN), cdivu(M, BM), batch);
  k_gemm_bf16_wmma<<<g, TB, 0, st>>>(A, lda, sA, tA, Bw, ldb, sB, tB,
                                     D, Dbf, ldd, sD, storeT, bias, sBias,
                                     colscale, resid, sRes, alpha, act, M, N, Kd);
}

extern "C" void kernel_launch(void* const* d_in, const int* in_sizes, int n_in,
                              void* d_out, int out_size, void* d_ws, size_t ws_size,
                              hipStream_t stream) {
  (void)in_sizes; (void)n_in; (void)out_size; (void)ws_size;
  const int Bn = 32, Sn = 512, Cn = 321, HDn = 512, Kc = 512, PLn = 96, MLPn = 512, NBn = 4;

  const float* x     = (const float*)d_in[0];
  const float* E     = (const float*)d_in[1];
  const float* Wm1   = (const float*)d_in[2];
  const float* bm1   = (const float*)d_in[3];
  const float* Wm2   = (const float*)d_in[4];
  const float* bm2   = (const float*)d_in[5];
  const float* Wq    = (const float*)d_in[6];
  const float* bq    = (const float*)d_in[7];
  const float* Wk    = (const float*)d_in[8];
  const float* bk    = (const float*)d_in[9];
  const float* Wout  = (const float*)d_in[10];
  const float* bout  = (const float*)d_in[11];
  const float* rev_w = (const float*)d_in[12];
  const float* rev_b = (const float*)d_in[13];
  const float* bn_t_g= (const float*)d_in[14];
  const float* bn_t_b= (const float*)d_in[15];
  const float* Wt    = (const float*)d_in[16];
  const float* bt    = (const float*)d_in[17];
  const float* bn_f_g= (const float*)d_in[18];
  const float* bn_f_b= (const float*)d_in[19];
  const float* Wf1   = (const float*)d_in[20];
  const float* bf1   = (const float*)d_in[21];
  const float* Wf2   = (const float*)d_in[22];
  const float* bf2   = (const float*)d_in[23];
  const float* Wol   = (const float*)d_in[24];
  const float* bol   = (const float*)d_in[25];

  float* out0  = (float*)d_out;                       // [B,PL,C]
  float* outCm = out0 + (long long)Bn * PLn * Cn;     // [B,K,HD]

  // bump allocator over d_ws
  char* wp = (char*)d_ws;
  auto alloc = [&](size_t bytes) -> void* {
    void* p = wp; wp += (bytes + 255) & ~(size_t)255; return p;
  };
  auto abf = [&](size_t nelem) -> unsigned short* { return (unsigned short*)alloc(nelem * 2); };
  auto af  = [&](size_t nelem) -> float* { return (float*)alloc(nelem * 4); };

  unsigned short* wWm1 = abf((size_t)HDn * Cn);
  unsigned short* wWm2 = abf((size_t)HDn * HDn);
  unsigned short* wWq  = abf((size_t)HDn * HDn);
  unsigned short* wWk  = abf((size_t)HDn * HDn);
  unsigned short* wE   = abf((size_t)Kc * HDn);
  unsigned short* wWt  = abf((size_t)Sn * Sn);
  unsigned short* wWf1 = abf((size_t)MLPn * Cn);
  unsigned short* wWf2 = abf((size_t)Cn * MLPn);
  unsigned short* wWol = abf((size_t)Cn * Sn * PLn);
  unsigned short* xn   = abf((size_t)Bn * Sn * Cn);
  unsigned short* T1   = abf((size_t)Bn * Sn * HDn);
  unsigned short* Hb   = abf((size_t)Bn * Sn * HDn);
  unsigned short* Pbf  = abf((size_t)Bn * Sn * Kc);
  unsigned short* Ktb  = abf((size_t)Bn * Sn * HDn);
  unsigned short* Qb   = abf((size_t)Kc * HDn);
  unsigned short* Abf  = abf((size_t)Bn * Kc * Sn);
  unsigned short* thWb = abf((size_t)Bn * Cn * HDn);
  unsigned short* ybn  = abf((size_t)Bn * Sn * Cn);
  unsigned short* Hmlp = abf((size_t)Bn * Sn * MLPn);
  unsigned short* Ztb  = abf((size_t)Bn * Cn * Sn);

  float* F     = af((size_t)Bn * Kc * Sn);   // reused: logits -> scores -> Cmraw -> Y
  float* theta = af((size_t)Bn * Cn * HDn);
  float* Zf    = af((size_t)Bn * Sn * Cn);
  float* muB   = af((size_t)Bn * Cn);
  float* sdB   = af((size_t)Bn * Cn);
  float* invE  = af((size_t)Kc);
  float* stm   = af((size_t)Sn);
  float* sti   = af((size_t)Sn);
  float* cfm   = af((size_t)Cn);
  float* cfi   = af((size_t)Cn);
  float* Otmp  = af((size_t)Cn * Bn * PLn);

  auto conv = [&](const float* src, unsigned short* dst, long long nelem) {
    k_f32_to_bf16<<<cdivu(nelem, 256), 256, 0, stream>>>(src, dst, nelem);
  };
  conv(Wm1, wWm1, (long long)HDn * Cn);
  conv(Wm2, wWm2, (long long)HDn * HDn);
  conv(Wq,  wWq,  (long long)HDn * HDn);
  conv(Wk,  wWk,  (long long)HDn * HDn);
  conv(E,   wE,   (long long)Kc * HDn);
  conv(Wt,  wWt,  (long long)Sn * Sn);
  conv(Wf1, wWf1, (long long)MLPn * Cn);
  conv(Wf2, wWf2, (long long)Cn * MLPn);
  conv(Wol, wWol, (long long)Cn * Sn * PLn);

  k_row_invnorm<<<Kc, TB, 0, stream>>>(E, invE, HDn);
  k_l2norm_rows_bf16<<<Bn * Sn, TB, 0, stream>>>(x, xn, Cn);

  const long long BSC = (long long)Bn * Sn * Cn;
  const long long SH  = (long long)Sn * HDn;
  const long long KS  = (long long)Kc * Sn;
  const long long CH  = (long long)Cn * HDn;
  const long long SC  = (long long)Sn * Cn;

  // ---- CCM ----
  // T1 = relu(xn @ Wm1^T + bm1)
  gemm(stream, xn, Cn, 0, 0, wWm1, Cn, 0, 1, nullptr, T1, HDn, 0, 0,
       bm1, 0, nullptr, nullptr, 0, 1.0f, 1, Bn * Sn, HDn, Cn, 1);
  // H = T1 @ Wm2^T + bm2
  gemm(stream, T1, HDn, 0, 0, wWm2, HDn, 0, 1, nullptr, Hb, HDn, 0, 0,
       bm2, 0, nullptr, nullptr, 0, 1.0f, 0, Bn * Sn, HDn, HDn, 1);
  // logits = (H @ E^T) * 1/||E_k||   -> F
  gemm(stream, Hb, HDn, 0, 0, wE, HDn, 0, 1, F, nullptr, Kc, 0, 0,
       nullptr, 0, invE, nullptr, 0, 1.0f, 0, Bn * Sn, Kc, HDn, 1);
  // P = softmax(logits)
  k_softmax_rows_bf16<<<Bn * Sn, TB, 0, stream>>>(F, Pbf, Kc);
  // Q = E @ Wq^T + bq
  gemm(stream, wE, HDn, 0, 0, wWq, HDn, 0, 1, nullptr, Qb, HDn, 0, 0,
       bq, 0, nullptr, nullptr, 0, 1.0f, 0, Kc, HDn, HDn, 1);
  // Kt = H @ Wk^T + bk
  gemm(stream, Hb, HDn, 0, 0, wWk, HDn, 0, 1, nullptr, Ktb, HDn, 0, 0,
       bk, 0, nullptr, nullptr, 0, 1.0f, 0, Bn * Sn, HDn, HDn, 1);
  // scores = Q @ Kt^T / sqrt(HD)  (batched over b) -> F
  gemm(stream, Qb, HDn, 0, 0, Ktb, HDn, SH, 1, F, nullptr, Sn, KS, 0,
       nullptr, 0, nullptr, nullptr, 0, 0.044194173824159216f, 0, Kc, Sn, HDn, Bn);
  // A = softmax(scores)
  k_softmax_rows_bf16<<<Bn * Kc, TB, 0, stream>>>(F, Abf, Sn);
  // Cmraw = A @ H  -> F
  gemm(stream, Abf, Sn, KS, 0, Hb, HDn, SH, 0, F, nullptr, HDn, (long long)Kc * HDn, 0,
       nullptr, 0, nullptr, nullptr, 0, 1.0f, 0, Kc, HDn, Sn, Bn);
  // Cm = l2norm(Cmraw * mask)  -> d_out
  k_cm_mask_l2norm<<<Bn * Kc, TB, 0, stream>>>(F, Pbf, outCm, Kc, Sn, HDn);
  // theta = P[:, :, :C]^T @ E  (batched over b)
  gemm(stream, Pbf, Kc, (long long)Sn * Kc, 1, wE, HDn, 0, 0, theta, nullptr, HDn, CH, 0,
       nullptr, 0, nullptr, nullptr, 0, 1.0f, 0, Cn, HDn, Sn, Bn);
  // thetaW = theta * Wout (broadcast over b), bf16
  k_mul_wout_bf16<<<cdivu((long long)Bn * CH, 256), 256, 0, stream>>>(
      theta, Wout, thWb, Cn, HDn, (long long)Bn * CH);
  // Y = H @ thetaW^T + bout  -> F
  gemm(stream, Hb, HDn, SH, 0, thWb, HDn, CH, 1, F, nullptr, Cn, SC, 0,
       bout, 0, nullptr, nullptr, 0, 1.0f, 0, Sn, Cn, HDn, Bn);

  // ---- RevIN norm ----
  k_revin_stats<<<cdivu(Bn * Cn, 256), 256, 0, stream>>>(F, muB, sdB, Bn, Sn, Cn);
  k_revin_apply<<<cdivu(BSC, 256), 256, 0, stream>>>(F, muB, sdB, rev_w, rev_b, Zf, Sn, Cn, BSC);

  // ---- Mixer blocks (shared weights, NB iterations) ----
  for (int it = 0; it < NBn; ++it) {
    // token mixing: Z += gelu(bn_t(Z) swap @ Wt^T + bt) swap
    k_bn_stats_s<<<Sn, TB, 0, stream>>>(Zf, stm, sti, Bn, Sn, Cn);
    k_bn_apply_s<<<cdivu(BSC, 256), 256, 0, stream>>>(Zf, stm, sti, bn_t_g, bn_t_b,
                                                      ybn, Sn, Cn, BSC);
    gemm(stream, ybn, Cn, SC, 1, wWt, Sn, 0, 1, Zf, nullptr, Cn, SC, 1,
         bt, 0, nullptr, Zf, SC, 1.0f, 2, Cn, Sn, Sn, Bn);
    // channel MLP: Z += gelu(bn_f(Z) @ Wf1^T + bf1) @ Wf2^T + bf2
    k_bn_stats_c<<<Cn, TB, 0, stream>>>(Zf, cfm, cfi, Bn, Sn, Cn);
    k_bn_apply_c<<<cdivu(BSC, 256), 256, 0, stream>>>(Zf, cfm, cfi, bn_f_g, bn_f_b,
                                                      ybn, Sn, Cn, BSC);
    gemm(stream, ybn, Cn, 0, 0, wWf1, Cn, 0, 1, nullptr, Hmlp, MLPn, 0, 0,
         bf1, 0, nullptr, nullptr, 0, 1.0f, 2, Bn * Sn, MLPn, Cn, 1);
    gemm(stream, Hmlp, MLPn, 0, 0, wWf2, MLPn, 0, 1, Zf, nullptr, Cn, 0, 0,
         bf2, 0, nullptr, Zf, 0, 1.0f, 0, Bn * Sn, Cn, MLPn, 1);
  }

  // ---- per-channel output heads: out[b,c,p] = sum_s Z[b,s,c]*Wol[c,s,p] + bol[c,p]
  k_transpose_bsc_bcs<<<cdivu(BSC, 256), 256, 0, stream>>>(Zf, Ztb, Sn, Cn, BSC);
  gemm(stream, Ztb, Cn * Sn, Sn, 0, wWol, PLn, (long long)Sn * PLn, 0,
       Otmp, nullptr, PLn, (long long)Bn * PLn, 0,
       bol, PLn, nullptr, nullptr, 0, 1.0f, 0, Bn, PLn, Sn, Cn);

  // ---- RevIN denorm -> d_out [B,PL,C]
  k_denorm_out<<<cdivu((long long)Bn * PLn * Cn, 256), 256, 0, stream>>>(
      Otmp, rev_w, rev_b, muB, sdB, out0, Bn, PLn, Cn);
}
